// StageNet_9199819948521
// MI455X (gfx1250) — matrix operand order repaired and checked by reading, not compiled, not run on hardware
//
#include <hip/hip_runtime.h>
#include <math.h>

#define HWC   20480   // H*W
#define WD    160
#define HD    128
#define DD    32
#define CC    32
#define NSRC  4
#define NVIEWS 5

typedef __attribute__((ext_vector_type(16))) _Float16 v16h;
typedef __attribute__((ext_vector_type(2)))  __fp16   fp16x2;
typedef __attribute__((ext_vector_type(8)))  float    v8f;
typedef __attribute__((ext_vector_type(8)))  unsigned v8u;
typedef __attribute__((ext_vector_type(2)))  float    f2;

__device__ __forceinline__ unsigned pkh2(float a, float b) {
  fp16x2 r = __builtin_amdgcn_cvt_pkrtz(a, b);   // v_cvt_pkrtz_f16_f32
  return __builtin_bit_cast(unsigned, r);
}
__device__ __forceinline__ float wred_sum(float v) {
#pragma unroll
  for (int o = 16; o > 0; o >>= 1) v += __shfl_xor(v, o, 32);
  return v;
}
__device__ __forceinline__ float wred_max(float v) {
#pragma unroll
  for (int o = 16; o > 0; o >>= 1) v = fmaxf(v, __shfl_xor(v, o, 32));
  return v;
}
// softmax over the 32 lanes; inputs here are bounded (|x| << 80) so the
// max-subtraction pass is dropped (shift-invariant, saves 5 serialized
// ds_bpermute+wait chains per call)
__device__ __forceinline__ float wsoftmax32(float x) {
  float e = __expf(x);
  float s = wred_sum(e);
  return e / s;
}

// ---------------- setup: per-view warp matrices into d_ws ----------------
__device__ void compose4(const float* p, float* o) {
  const float* E = p;       // extrinsic p[:,0]
  const float* K = p + 16;  // intrinsic p[:,1]
#pragma unroll
  for (int i = 0; i < 16; i++) o[i] = E[i];
  for (int i = 0; i < 3; i++)
    for (int j = 0; j < 4; j++) {
      float a = 0.f;
      for (int k = 0; k < 3; k++) a += K[4 * i + k] * E[4 * k + j];
      o[4 * i + j] = a;
    }
}
__device__ void inv4(const float* A, float* o) {
  float M[4][8];
  for (int i = 0; i < 4; i++)
    for (int j = 0; j < 4; j++) { M[i][j] = A[4 * i + j]; M[i][4 + j] = (i == j) ? 1.f : 0.f; }
  for (int c = 0; c < 4; c++) {
    int piv = c; float best = fabsf(M[c][c]);
    for (int r = c + 1; r < 4; r++) { float f = fabsf(M[r][c]); if (f > best) { best = f; piv = r; } }
    if (piv != c)
      for (int j = 0; j < 8; j++) { float t = M[c][j]; M[c][j] = M[piv][j]; M[piv][j] = t; }
    float d = 1.f / M[c][c];
    for (int j = 0; j < 8; j++) M[c][j] *= d;
    for (int r = 0; r < 4; r++) {
      if (r == c) continue;
      float f = M[r][c];
      for (int j = 0; j < 8; j++) M[r][j] = fmaf(-f, M[c][j], M[r][j]);
    }
  }
  for (int i = 0; i < 4; i++)
    for (int j = 0; j < 4; j++) o[4 * i + j] = M[i][4 + j];
}
__global__ void setup_mats(const float* __restrict__ proj, float* __restrict__ ws) {
  if (threadIdx.x != 0 || blockIdx.x != 0) return;
  float refm[16], invref[16];
  compose4(proj, refm);          // view 0
  inv4(refm, invref);
  for (int v = 0; v < NSRC; ++v) {
    float P[16], M[16];
    compose4(proj + (size_t)(v + 1) * 32, P);
    for (int i = 0; i < 4; i++)
      for (int j = 0; j < 4; j++) {
        float a = 0.f;
        for (int k = 0; k < 4; k++) a += P[4 * i + k] * invref[4 * k + j];
        M[4 * i + j] = a;
      }
    float* o = ws + v * 12;
    o[0] = M[0]; o[1] = M[1]; o[2] = M[2];
    o[3] = M[4]; o[4] = M[5]; o[5] = M[6];
    o[6] = M[8]; o[7] = M[9]; o[8] = M[10];
    o[9] = M[3]; o[10] = M[7]; o[11] = M[11];
  }
}

// ---------------- fused main kernel: wave = pixel, lane = depth ----------------
__global__ __launch_bounds__(256) void stage_fused(
    const float* __restrict__ ref_feat,
    const float* __restrict__ src_feats,
    const float* __restrict__ depth_hypo,
    const float* __restrict__ w_reg,
    const float* __restrict__ w_var,
    const float* __restrict__ mats,
    float* __restrict__ out)
{
  const int lane = threadIdx.x & 31;
  const int wid  = threadIdx.x >> 5;
  const int p    = blockIdx.x * 8 + wid;      // pixel id (exact cover)
  const int m    = lane & 15;
  const int hh   = lane >> 4;

  const float xf = (float)(p % WD);
  const float yf = (float)(p / WD);
  const float depth = depth_hypo[lane * HWC + p];

  // per-lane (=per depth) accumulators
  float sum[CC];           // running Σ_views warped[c] (incl. ref)
  float wv[CC];            // w_var
  float wsq = 0.f;         // Σ_views Σ_c w_var[c] * warped[c]^2 (incl. ref)
  const float* refp = ref_feat + p;
#pragma unroll
  for (int c = 0; c < CC; c++) {
    float r = refp[c * HWC];
    wv[c]  = w_var[c];
    sum[c] = r;
    wsq = fmaf(wv[c] * r, r, wsq);
  }

  // WMMA A-matrix: 16x32 f16, rows g=0..7 = block-diagonal ref/4 (group-mean)
  v8u au;
#pragma unroll
  for (int v = 0; v < 8; ++v) {
    int k0 = 2 * v + 16 * hh;
    float a0 = 0.f, a1 = 0.f;
    if (m < 8) {
      int lo = 4 * m;
      if (k0     >= lo && k0     < lo + 4) a0 = 0.25f * refp[k0 * HWC];
      if (k0 + 1 >= lo && k0 + 1 < lo + 4) a1 = 0.25f * refp[(k0 + 1) * HWC];
    }
    au[v] = pkh2(a0, a1);
  }
  const v16h Amat = __builtin_bit_cast(v16h, au);

  float corf[8];
#pragma unroll
  for (int g = 0; g < 8; g++) corf[g] = 0.f;
  float cwsum = 1e-8f;
  const float inv_sqrt_c = 0.17677669529663687f;  // 1/sqrt(32)

  for (int v = 0; v < NSRC; ++v) {
    const float* Rm = mats + v * 12;
    float r0 = Rm[0], r1 = Rm[1], r2 = Rm[2];
    float r3 = Rm[3], r4 = Rm[4], r5 = Rm[5];
    float r6 = Rm[6], r7 = Rm[7], r8 = Rm[8];
    float t0 = Rm[9], t1 = Rm[10], t2 = Rm[11];

    float rx = fmaf(r0, xf, fmaf(r1, yf, r2));
    float ry = fmaf(r3, xf, fmaf(r4, yf, r5));
    float rz = fmaf(r6, xf, fmaf(r7, yf, r8));
    float X = fmaf(rx, depth, t0);
    float Y = fmaf(ry, depth, t1);
    float Z = fmaf(rz, depth, t2);
    float px = X / Z, py = Y / Z;

    float x0f = floorf(px), y0f = floorf(py);
    float wx1 = px - x0f, wx0 = 1.f - wx1;
    float wy1 = py - y0f, wy0 = 1.f - wy1;
    bool vx0 = (x0f >= 0.f) && (x0f <= (float)(WD - 1));
    bool vx1 = (x0f + 1.f >= 0.f) && (x0f + 1.f <= (float)(WD - 1));
    bool vy0 = (y0f >= 0.f) && (y0f <= (float)(HD - 1));
    bool vy1 = (y0f + 1.f >= 0.f) && (y0f + 1.f <= (float)(HD - 1));
    int x0i = (int)x0f, y0i = (int)y0f;
    int x0c = min(max(x0i, 0), WD - 1), x1c = min(max(x0i + 1, 0), WD - 1);
    int y0c = min(max(y0i, 0), HD - 1), y1c = min(max(y0i + 1, 0), HD - 1);
    float w00 = (vx0 && vy0) ? wx0 * wy0 : 0.f;
    float w10 = (vx1 && vy0) ? wx1 * wy0 : 0.f;
    float w01 = (vx0 && vy1) ? wx0 * wy1 : 0.f;
    float w11 = (vx1 && vy1) ? wx1 * wy1 : 0.f;

    // Both x-taps lie inside {xb, xb+1} with xb = min(x0c, W-2), so each
    // bilinear row is ONE b64 load. Remap the 4 weights onto the vector
    // slots once per view (exact for interior / left-clamp / right-clamp).
    int  xb   = min(x0c, WD - 2);
    bool s0hi = (x0c > xb);          // x0 tap sits in .y slot (right edge)
    bool s1hi = (x1c > xb);          // x1 tap sits in .y slot (interior/right)
    float wa0 = (s0hi ? 0.f : w00) + (s1hi ? 0.f : w10);
    float wb0 = (s0hi ? w00 : 0.f) + (s1hi ? w10 : 0.f);
    float wa1 = (s0hi ? 0.f : w01) + (s1hi ? 0.f : w11);
    float wb1 = (s0hi ? w01 : 0.f) + (s1hi ? w11 : 0.f);
    int r0off = y0c * WD + xb;
    int r1off = y1c * WD + xb;

    const float* sf = src_feats + (size_t)v * CC * HWC;
    float warped[CC];
#pragma unroll
    for (int c = 0; c < CC; c++) {
      const float* fc = sf + c * HWC;
      f2 a = *(const f2*)(fc + r0off);   // 4B-aligned b64: OK in DWORD align mode
      f2 b = *(const f2*)(fc + r1off);
      float acc = a.x * wa0;
      acc = fmaf(a.y, wb0, acc);
      acc = fmaf(b.x, wa1, acc);
      acc = fmaf(b.y, wb1, acc);
      warped[c] = acc;
      sum[c] += acc;
      wsq = fmaf(wv[c] * acc, acc, wsq);
    }

    // --- correlation via V_WMMA_F32_16X16X32_F16 ---
    // B (32x16) layout used here (contiguous k, same as A): VGPR k: lanes 0-15 hold K=2k,2k+1 ; lanes 16-31 hold K=16+2k.
    unsigned pk[16];
#pragma unroll
    for (int j = 0; j < 16; j++) pk[j] = pkh2(warped[2 * j], warped[2 * j + 1]);

    v8u blo, bhi;
#pragma unroll
    for (int k = 0; k < 8; k++) {
      unsigned up = (unsigned)__shfl((int)pk[8 + k], m, 32);       // lanes16-31 <- lane m
      blo[k] = hh ? up : pk[k];
      unsigned dn = (unsigned)__shfl((int)pk[k], m + 16, 32);      // lanes0-15 <- lane m+16
      bhi[k] = hh ? pk[8 + k] : dn;
    }
    v8f cz = {};
    v8f c0 = __builtin_amdgcn_wmma_f32_16x16x32_f16(
        false, Amat, false, __builtin_bit_cast(v16h, blo), (short)0, cz, false, false);
    v8f c1 = __builtin_amdgcn_wmma_f32_16x16x32_f16(
        false, Amat, false, __builtin_bit_cast(v16h, bhi), (short)0, cz, false, false);

    // cor[g] for this lane's depth
    float logit = 0.f;
    float cor[8];
#pragma unroll
    for (int g = 0; g < 8; ++g) {
      float ot = __shfl(c1[g], m, 32);
      cor[g] = hh ? ot : c0[g];
      logit += cor[g];
    }
    logit *= 0.5f;                      // / ATTN_TEMP

    // depth-softmax attention weight (cross-lane, no max pass needed)
    float e  = __expf(logit);
    float s  = wred_sum(e);
    float cw = (e / s) * inv_sqrt_c;
    cwsum += cw;
#pragma unroll
    for (int g = 0; g < 8; ++g) corf[g] = fmaf(cw, cor[g], corf[g]);
  }

  // ---- finalize ----
  float inv_cws = 1.f / cwsum;
  float cost = 0.f;
#pragma unroll
  for (int g = 0; g < 8; g++) cost = fmaf(corf[g] * inv_cws, w_reg[g], cost);

  // var_cost = Σ_c w_var[c]*(sq_sum/5 - (sum/5)^2)
  float s2 = 0.f;
#pragma unroll
  for (int c = 0; c < CC; c++) { float sm = sum[c] * 0.2f; s2 = fmaf(wv[c] * sm, sm, s2); }
  float varcost = wsq * 0.2f - s2;

  float sim_p = wsoftmax32(cost);
  float var_p = wsoftmax32(varcost);
  float pv    = wsoftmax32(sim_p * var_p);

  out[4 * HWC + lane * HWC + p] = pv;   // prob_volume

  // argmax over depth (first max on ties, matching jnp.argmax)
  float bv = pv; int bi = lane;
#pragma unroll
  for (int o = 16; o > 0; o >>= 1) {
    float ov = __shfl_xor(bv, o, 32);
    int   oi = __shfl_xor(bi, o, 32);
    if (ov > bv || (ov == bv && oi < bi)) { bv = ov; bi = oi; }
  }
  float dbest = __shfl(depth, bi, 32);
  float vconf = wred_max(var_p);
  float sconf = wred_max(sim_p);
  float d1 = __shfl(depth, 1, 32);
  float d2 = __shfl(depth, 2, 32);
  float itv = 1.f / d2 - 1.f / d1;

  if (lane == 0) {
    out[p]            = dbest;                      // depth
    out[HWC + p]      = bv;                         // conf
    out[2 * HWC + p]  = vconf;                      // var_conf
    out[3 * HWC + p]  = sconf;                      // sim_conf
    out[36 * HWC + p] = itv;                        // last_depth_itv
    out[37 * HWC + p] = 1.f / dbest + 0.5f * itv;   // inverse_min_depth
    out[38 * HWC + p] = 1.f / dbest - 0.5f * itv;   // inverse_max_depth
  }
}

extern "C" void kernel_launch(void* const* d_in, const int* in_sizes, int n_in,
                              void* d_out, int out_size, void* d_ws, size_t ws_size,
                              hipStream_t stream) {
  const float* ref_feat   = (const float*)d_in[0];
  const float* src_feats  = (const float*)d_in[1];
  const float* proj       = (const float*)d_in[2];
  const float* depth_hypo = (const float*)d_in[3];
  const float* w_reg      = (const float*)d_in[4];
  const float* w_var      = (const float*)d_in[5];
  // d_in[6] = text_mask (unused by the reference)
  float* out  = (float*)d_out;
  float* mats = (float*)d_ws;   // 4 views * 12 floats = 192 bytes

  setup_mats<<<1, 32, 0, stream>>>(proj, mats);
  stage_fused<<<HWC / 8, 256, 0, stream>>>(ref_feat, src_feats, depth_hypo,
                                           w_reg, w_var, mats, out);
}
